// MyModel_58196806861158
// MI455X (gfx1250) — matrix operand order repaired and checked by reading, not compiled, not run on hardware
//
#include <hip/hip_runtime.h>
#include <hip/hip_bf16.h>
#include <math.h>

// ---------------------------------------------------------------------------
// MI455X (gfx1250) fused MLP + reductions.
// Roofline: ~600MB HBM traffic @ 23.3 TB/s => ~26us floor; ~39 GFLOP of GEMM
// done as f16 WMMA (f32 accum) => well under memory time. So: one streaming
// pass over x/h, all 6 layers per 16-row tile via v_wmma_f32_16x16x32_f16,
// weights staged once as padded f16 in d_ws (L2-resident), reductions fused.
// ---------------------------------------------------------------------------

typedef __attribute__((ext_vector_type(16))) _Float16 v16h;
typedef __attribute__((ext_vector_type(8)))  _Float16 h8;
typedef __attribute__((ext_vector_type(8)))  float    v8f;

#define NROWS   500000
#define NT_TILES 31250          // NROWS / 16
#define WAVES   4               // waves per block, 1 tile per wave
#define NBLOCKS ((NT_TILES + WAVES - 1) / WAVES)   // 7813
#define NPART   103             // per-block partials: s0, s2, s3, colsum[100]

// d_ws layout.  Halved (f16) weight offsets, padded [Npad][Kpad], zero-filled:
#define W1_OFF  0               // 112 x 224
#define W12_OFF 25088           //  64 x 128
#define W13_OFF 33280           //  32 x 160
#define W14_OFF 38400           //  32 x  32
#define W15_OFF 39424           //  48 x 128
#define W2_OFF  45568           // 112 x  64
#define WTOT    52736
#define BIAS_BYTE_OFF (WTOT * 2)            // padded f32 biases
#define B1_OFF  0
#define B12_OFF 112
#define B13_OFF 176
#define B14_OFF 208
#define B15_OFF 240
#define B2_OFF  288
#define BTOT    400
#define PART_BYTE_OFF (BIAS_BYTE_OFF + BTOT * 4)   // block partials (f32)

// ---------------------------------------------------------------------------
// Stage weights: f32 [Nout,Kin] -> f16 [Npad,Kpad] zero-padded; bias padded f32.
// ---------------------------------------------------------------------------
__global__ void prep_weights(const float* __restrict__ w, const float* __restrict__ b,
                             int Nout, int Kin, int Npad, int Kpad,
                             _Float16* __restrict__ dw, float* __restrict__ db) {
    int i = blockIdx.x * blockDim.x + threadIdx.x;
    int tot = Npad * Kpad;
    if (i < tot) {
        int n = i / Kpad, k = i % Kpad;
        dw[i] = (n < Nout && k < Kin) ? (_Float16)w[n * Kin + k] : (_Float16)0.0f;
    }
    if (i < Npad) db[i] = (i < Nout) ? b[i] : 0.0f;
}

// ---------------------------------------------------------------------------
// One layer: D[16xN] = relu(A[16xK] * W^T + bias), all tiles 16x16xK via WMMA.
// A fragments for the whole layer are hoisted to registers BEFORE any store,
// so src LDS == dst LDS is safe (per-wave LDS ops are in-order).
// A frag (16-bit 16x32 layout): lane holds row m=lane&15; elements 0-7 are
// K = kb + 8*hi + [0..8), elements 8-15 are K = kb + 16 + 8*hi + [0..8).
// B frag (32x16): lane holds col n=lane&15; same K order as A frag.
// ---------------------------------------------------------------------------
template <int KCH, int NTILES, bool RELU, bool F32OUT>
__device__ __forceinline__ void layer(const _Float16* __restrict__ src, int sstride,
                                      _Float16* __restrict__ dst, int dstride,
                                      float* __restrict__ fdst, int fstride,
                                      const _Float16* __restrict__ wg, int kpad,
                                      const float* __restrict__ bg, int lane) {
    const int ln = lane & 15, hi = lane >> 4;
    v16h A[KCH];
#pragma unroll
    for (int kc = 0; kc < KCH; ++kc) {
        const h8* plo = (const h8*)(src + ln * sstride + kc * 32 + 8 * hi);
        const h8* phi = (const h8*)(src + ln * sstride + kc * 32 + 16 + 8 * hi);
        h8 lo = *plo, hv = *phi;
        A[kc] = __builtin_shufflevector(lo, hv, 0, 1, 2, 3, 4, 5, 6, 7,
                                        8, 9, 10, 11, 12, 13, 14, 15);
    }
#pragma unroll
    for (int nt = 0; nt < NTILES; ++nt) {
        const int n = nt * 16 + ln;
        v8f acc = {};
#pragma unroll
        for (int kc = 0; kc < KCH; ++kc) {
            const h8* pb = (const h8*)(wg + (size_t)n * kpad + kc * 32 + 8 * hi);
            h8 b0 = pb[0], b1 = pb[2];
            v16h B = __builtin_shufflevector(b0, b1, 0, 1, 2, 3, 4, 5, 6, 7,
                                             8, 9, 10, 11, 12, 13, 14, 15);
            acc = __builtin_amdgcn_wmma_f32_16x16x32_f16(
                false, A[kc], false, B, (short)0, acc, false, false);
        }
        const float bias = bg[n];
#pragma unroll
        for (int r = 0; r < 8; ++r) {
            float v = acc[r] + bias;
            if (RELU) v = v > 0.0f ? v : 0.0f;
            const int m = r + 8 * hi;
            if constexpr (F32OUT)
                fdst[m * fstride + n] = v;
            else
                dst[m * dstride + n] = (_Float16)v;
        }
    }
}

// ---------------------------------------------------------------------------
// Main fused kernel: 4 waves/block, 1 tile (16 rows) per wave.
// ---------------------------------------------------------------------------
__global__ __launch_bounds__(128) void mlp_fused(
    const float* __restrict__ x, const float* __restrict__ h,
    const _Float16* __restrict__ wq, const float* __restrict__ bq,
    float* __restrict__ out, float* __restrict__ partials) {
    const int tid  = threadIdx.x;
    const int lane = tid & 31;
    const int wv   = tid >> 5;
    const int tile = blockIdx.x * WAVES + wv;
    const bool active = (tile < NT_TILES);   // wave-uniform

    // buf0: 16 x 232 halves (x|h|zero-pad, immutable after load)
    // bufI: 16 x 224 halves, in-place intermediate; aliased as f32 16 x 112 for f
    __shared__ __align__(16) _Float16 s0buf[WAVES][16 * 232];
    __shared__ __align__(16) _Float16 sibuf[WAVES][16 * 224];
    __shared__ float scol[100];
    __shared__ float sred[3][128];

    if (tid < 100) scol[tid] = 0.0f;
    __syncthreads();

    float a0 = 0.0f, a2 = 0.0f, a3 = 0.0f;

    if (active) {
        const int r0 = tile * 16;
        _Float16* b0 = s0buf[wv];
        _Float16* bi = sibuf[wv];
        float*    fb = (float*)bi;          // same bytes, row stride 448B both

        // ---- load x|h -> f16, fuse part1 (column sums) + part2 (sum h^2) ----
        for (int idx = lane; idx < 16 * 200; idx += 32) {
            const int r = idx / 200, c = idx % 200;
            const float v = (c < 100) ? x[(size_t)(r0 + r) * 100 + c]
                                      : h[(size_t)(r0 + r) * 100 + (c - 100)];
            b0[r * 232 + c] = (_Float16)v;
            if (c >= 100) {
                a2 += v * v;
                atomicAdd(&scol[c - 100], v);   // ds_add_f32
            }
        }
        for (int idx = lane; idx < 16 * 24; idx += 32) {    // K pad 200..223
            const int r = idx / 24, c = 200 + idx % 24;
            b0[r * 232 + c] = (_Float16)0.0f;
        }
        // ---- part3: sum |h[i+1]-h[i]| (next row from buf0, last from HBM) ----
        for (int idx = lane; idx < 16 * 100; idx += 32) {
            const int r = idx / 100, c = idx % 100;
            const int i = r0 + r;
            if (i < NROWS - 1) {
                const float cur = (float)b0[r * 232 + 100 + c];
                const float nxt = (r < 15) ? (float)b0[(r + 1) * 232 + 100 + c]
                                           : h[(size_t)(i + 1) * 100 + c];
                a3 += fabsf(nxt - cur);
            }
        }

        // ---- L1: [16x224]x[224x112] ----
        layer<7, 7, true, false>(b0, 232, bi, 224, nullptr, 0,
                                 wq + W1_OFF, 224, bq + B1_OFF, lane);
        for (int idx = lane; idx < 16 * 16; idx += 32) {    // zero 112..127
            const int r = idx / 16, c = 112 + idx % 16;
            bi[r * 224 + c] = (_Float16)0.0f;
        }
        // ---- L2: [16x128]x[128x64] ----
        layer<4, 4, true, false>(bi, 224, bi, 224, nullptr, 0,
                                 wq + W12_OFF, 128, bq + B12_OFF, lane);
        for (int idx = lane; idx < 16 * 100; idx += 32) {   // concat h -> 50..149
            const int r = idx / 100, c = idx % 100;
            bi[r * 224 + 50 + c] = b0[r * 232 + 100 + c];
        }
        for (int idx = lane; idx < 16 * 10; idx += 32) {    // zero 150..159
            const int r = idx / 10, c = 150 + idx % 10;
            bi[r * 224 + c] = (_Float16)0.0f;
        }
        // ---- L3: [16x160]x[160x32] ----
        layer<5, 2, true, false>(bi, 224, bi, 224, nullptr, 0,
                                 wq + W13_OFF, 160, bq + B13_OFF, lane);
        // ---- L4: [16x32]x[32x32] ----
        layer<1, 2, true, false>(bi, 224, bi, 224, nullptr, 0,
                                 wq + W14_OFF, 32, bq + B14_OFF, lane);
        for (int idx = lane; idx < 16 * 100; idx += 32) {   // concat h -> 20..119
            const int r = idx / 100, c = idx % 100;
            bi[r * 224 + 20 + c] = b0[r * 232 + 100 + c];
        }
        for (int idx = lane; idx < 16 * 8; idx += 32) {     // zero 120..127
            const int r = idx / 8, c = 120 + idx % 8;
            bi[r * 224 + c] = (_Float16)0.0f;
        }
        // ---- L5: [16x128]x[128x48] ----
        layer<4, 3, true, false>(bi, 224, bi, 224, nullptr, 0,
                                 wq + W15_OFF, 128, bq + B15_OFF, lane);
        for (int idx = lane; idx < 16 * 16; idx += 32) {    // zero 48..63
            const int r = idx / 16, c = 48 + idx % 16;
            bi[r * 224 + c] = (_Float16)0.0f;
        }
        // ---- L6 (no relu, f32 out in-place over bufI) ----
        layer<2, 7, false, true>(bi, 224, nullptr, 0, fb, 112,
                                 wq + W2_OFF, 64, bq + B2_OFF, lane);

        // ---- write f coalesced + fuse part0: ||f[i] - x[i+1]||^2 ----
        for (int idx = lane; idx < 16 * 100; idx += 32) {
            const int r = idx / 100, c = idx % 100;
            const int i = r0 + r;
            const float fv = fb[r * 112 + c];
            out[(size_t)i * 100 + c] = fv;
            if (i < NROWS - 1) {
                const float xn = (r < 15) ? (float)b0[(r + 1) * 232 + c]
                                          : x[(size_t)(i + 1) * 100 + c];
                const float d = fv - xn;
                a0 += d * d;
            }
        }
    }

    // ---- deterministic per-block partials ----
    sred[0][tid] = a0;
    sred[1][tid] = a2;
    sred[2][tid] = a3;
    __syncthreads();
    if (tid < 3) {
        float s = 0.0f;
        for (int j = 0; j < 128; ++j) s += sred[tid][j];
        partials[(size_t)blockIdx.x * NPART + tid] = s;
    }
    if (tid < 100) partials[(size_t)blockIdx.x * NPART + 3 + tid] = scol[tid];
}

// ---------------------------------------------------------------------------
// Final reduction over block partials (double accumulation, fixed order).
// ---------------------------------------------------------------------------
__global__ void finalize_parts(const float* __restrict__ partials, int nblocks,
                               float* __restrict__ out) {
    __shared__ double colsum[100];
    __shared__ double sums[3];
    const int tid = threadIdx.x;
    if (tid < 100) {
        double s = 0.0;
        for (int b = 0; b < nblocks; ++b)
            s += (double)partials[(size_t)b * NPART + 3 + tid];
        colsum[tid] = s;
    }
    if (tid >= 100 && tid < 103) {
        const int k = tid - 100;
        double s = 0.0;
        for (int b = 0; b < nblocks; ++b)
            s += (double)partials[(size_t)b * NPART + k];
        sums[k] = s;
    }
    __syncthreads();
    if (tid == 0) {
        const double S0 = sums[0], S2 = sums[1], S3 = sums[2];
        double p1 = 0.0;
        for (int c = 0; c < 100; ++c) p1 += fabs(colsum[c] / (double)NROWS);
        p1 /= 100.0;
        float* po = out + (size_t)NROWS * 100;
        po[0] = (float)(sqrt(S0) / (double)(NROWS - 1));
        po[1] = (float)p1;
        po[2] = (float)fabs(S2 / ((double)NROWS * 100.0) - 1.0);
        po[3] = (float)(S3 / ((double)(NROWS - 1) * 100.0));
    }
}

// ---------------------------------------------------------------------------
extern "C" void kernel_launch(void* const* d_in, const int* in_sizes, int n_in,
                              void* d_out, int out_size, void* d_ws, size_t ws_size,
                              hipStream_t stream) {
    (void)in_sizes; (void)n_in; (void)out_size; (void)ws_size;
    const float* x   = (const float*)d_in[0];
    const float* h   = (const float*)d_in[1];
    const float* w1  = (const float*)d_in[2];   const float* b1  = (const float*)d_in[3];
    const float* w12 = (const float*)d_in[4];   const float* b12 = (const float*)d_in[5];
    const float* w13 = (const float*)d_in[6];   const float* b13 = (const float*)d_in[7];
    const float* w14 = (const float*)d_in[8];   const float* b14 = (const float*)d_in[9];
    const float* w15 = (const float*)d_in[10];  const float* b15 = (const float*)d_in[11];
    const float* w2  = (const float*)d_in[12];  const float* b2  = (const float*)d_in[13];

    _Float16* wq = (_Float16*)d_ws;
    float*    bq = (float*)((char*)d_ws + BIAS_BYTE_OFF);
    float*    pp = (float*)((char*)d_ws + PART_BYTE_OFF);   // needs ~3.3 MB ws
    float*    out = (float*)d_out;

    // Stage weights (re-run every call: deterministic, cheap).
    auto prep = [&](const float* w, const float* b, int Nout, int Kin,
                    int Npad, int Kpad, int woff, int boff) {
        const int tot = Npad * Kpad;
        prep_weights<<<(tot + 255) / 256, 256, 0, stream>>>(
            w, b, Nout, Kin, Npad, Kpad, wq + woff, bq + boff);
    };
    prep(w1,  b1,  100, 200, 112, 224, W1_OFF,  B1_OFF);
    prep(w12, b12,  50, 100,  64, 128, W12_OFF, B12_OFF);
    prep(w13, b13,  30, 150,  32, 160, W13_OFF, B13_OFF);
    prep(w14, b14,  20,  30,  32,  32, W14_OFF, B14_OFF);
    prep(w15, b15,  40, 120,  48, 128, W15_OFF, B15_OFF);
    prep(w2,  b2,  100,  40, 112,  64, W2_OFF,  B2_OFF);

    mlp_fused<<<NBLOCKS, 128, 0, stream>>>(x, h, wq, bq, out, pp);
    finalize_parts<<<1, 128, 0, stream>>>(pp, NBLOCKS, out);
}
